// Isotropic_VIG_lorentz_complete_42580305773099
// MI455X (gfx1250) — compile-verified, matrix operand order checked by reading, not run on hardware
//
#include <hip/hip_runtime.h>
#include <cstdint>
#include <cstddef>

typedef __attribute__((ext_vector_type(16))) _Float16 v16h;
typedef __attribute__((ext_vector_type(8)))  _Float16 v8h;
typedef __attribute__((ext_vector_type(8)))  float    v8f;

#define BN_INV 0.9999950000374997f

__device__ inline float waveRedAdd(float v) {
#pragma unroll
  for (int off = 16; off > 0; off >>= 1) v += __shfl_xor(v, off, 32);
  return v;
}

// ---------------------------------------------------------------------------
// f32 -> f16 convert (optionally fused ReLU, since lorentz_linear matmuls use relu(x))
__global__ void cvt_f16_k(const float* __restrict__ src, _Float16* __restrict__ dst,
                          size_t n, int relu) {
  size_t i = (size_t)blockIdx.x * blockDim.x + threadIdx.x;
  if (i >= n) return;
  float v = src[i];
  if (relu) v = fmaxf(v, 0.f);
  dst[i] = (_Float16)v;
}

__global__ void add_inplace_k(float* __restrict__ a, const float* __restrict__ b, size_t n) {
  size_t i = (size_t)blockIdx.x * blockDim.x + threadIdx.x;
  if (i >= n) return;
  a[i] += b[i];
}

// ---------------------------------------------------------------------------
// WMMA GEMM: Out[m,n] = f( sum_k A[m,k]*Bw[n,k] )
//   mode 0: Out = acc + bias[n]
//   mode 1: Out = sqv[m] + sqv[n] - 2*acc      (pairwise squared distances)
// A: M x K f16 row-major, Bw: Nn x K f16 row-major. M,Nn % 16 == 0, K % 32 == 0.
// One 16x16 tile per wave32, v_wmma_f32_16x16x32_f16, f32 accumulation.
__global__ void gemm_wmma_f16(const _Float16* __restrict__ A,
                              const _Float16* __restrict__ Bw,
                              const float* __restrict__ bias,
                              const float* __restrict__ sqv,
                              float* __restrict__ Out,
                              int M, int Nn, int K, int mode, int tiles) {
  int wv   = threadIdx.x >> 5;
  int lane = threadIdx.x & 31;
  int tile = blockIdx.x * 8 + wv;
  if (tile >= tiles) return;
  int tilesN = Nn >> 4;
  int tm = tile / tilesN;
  int tn = tile - tm * tilesN;
  int r    = lane & 15;
  int half = lane >> 4;
  // A fragment: lanes 0-15 hold row M=r with K 0..7 / 16..23; lanes 16-31 hold K 8..15 / 24..31
  const _Float16* arow = A  + (size_t)(tm * 16 + r) * K + half * 8;
  const _Float16* brow = Bw + (size_t)(tn * 16 + r) * K + half * 8;
  v8f acc = {};
  union U { v16h v; v8h h[2]; };
  for (int k0 = 0; k0 < K; k0 += 32) {
    U a, b;
    a.h[0] = *reinterpret_cast<const v8h*>(arow + k0);
    a.h[1] = *reinterpret_cast<const v8h*>(arow + k0 + 16);
    b.h[0] = *reinterpret_cast<const v8h*>(brow + k0);
    b.h[1] = *reinterpret_cast<const v8h*>(brow + k0 + 16);
    acc = __builtin_amdgcn_wmma_f32_16x16x32_f16(false, a.v, false, b.v,
                                                 (short)0, acc, false, false);
  }
  int n = tn * 16 + r;
#pragma unroll
  for (int i = 0; i < 8; ++i) {
    int m = tm * 16 + i + half * 8;   // C/D layout: VGPR i -> M=i (+8 for upper half-wave)
    float v = acc[i];
    if (mode == 0) v += bias[n];
    else           v = sqv[m] + sqv[n] - 2.f * v;
    Out[(size_t)m * Nn + n] = v;
  }
}

// ---------------------------------------------------------------------------
// Lorentz-linear epilogue, one wave per row (in place):
//   time = sigmoid(y0)*exp(log_s)+1.1 ; sc=(time^2-1)/max(sum(y[1:]^2),1e-8)
//   y = [time, y[1:]*sqrt(sc)]
__global__ void lorentz_rows_k(float* __restrict__ Y, int rows, int C,
                               const float* __restrict__ logs) {
  int wv = threadIdx.x >> 5, lane = threadIdx.x & 31;
  int row = blockIdx.x * 8 + wv;
  if (row >= rows) return;
  float* y = Y + (size_t)row * C;
  float s2 = 0.f;
  for (int c = lane; c < C; c += 32) {
    float v = y[c];
    if (c > 0) s2 += v * v;
  }
  s2 = waveRedAdd(s2);
  float y0 = y[0];
  float t  = 1.f / (1.f + expf(-y0)) * expf(logs[0]) + 1.1f;
  float sc = (t * t - 1.f) / fmaxf(s2, 1e-8f);
  float rt = sqrtf(sc);
  for (int c = lane; c < C; c += 32) y[c] = (c == 0) ? t : y[c] * rt;
}

// row-wise squared norm over all C channels (for the distance matrix)
__global__ void row_sqnorm_k(const float* __restrict__ f, float* __restrict__ sq,
                             int rows, int C) {
  int wv = threadIdx.x >> 5, lane = threadIdx.x & 31;
  int row = blockIdx.x * 8 + wv;
  if (row >= rows) return;
  const float* fr = f + (size_t)row * C;
  float s = 0.f;
  for (int c = lane; c < C; c += 32) { float v = fr[c]; s += v * v; }
  s = waveRedAdd(s);
  if (lane == 0) sq[row] = s;
}

// ---------------------------------------------------------------------------
// Top-k per distance row: keep KD=k*dil smallest (ascending), emit every dil-th index.
// Register-resident sorted insertion per lane, 32-way LDS merge by lane 0.
template <int KD>
__global__ void topk_rows_k(const float* __restrict__ dist, int* __restrict__ idxout,
                            int rows, int N, int dil, int k) {
  __shared__ float sv[8][32 * KD];
  __shared__ int   si[8][32 * KD];
  int wv = threadIdx.x >> 5, lane = threadIdx.x & 31;
  int row = blockIdx.x * 8 + wv;
  bool active = row < rows;
  float lv[KD]; int li[KD];
#pragma unroll
  for (int t = 0; t < KD; ++t) { lv[t] = 3.0e38f; li[t] = 0; }
  if (active) {
    const float* dr = dist + (size_t)row * N;
    for (int j = lane; j < N; j += 32) {
      float d = dr[j];
      if (d < lv[KD - 1]) {
        lv[KD - 1] = d; li[KD - 1] = j;
#pragma unroll
        for (int p = KD - 1; p > 0; --p) {  // single bubble pass = sorted insert
          if (lv[p] < lv[p - 1]) {
            float tv = lv[p]; lv[p] = lv[p - 1]; lv[p - 1] = tv;
            int   ti = li[p]; li[p] = li[p - 1]; li[p - 1] = ti;
          }
        }
      }
    }
#pragma unroll
    for (int t = 0; t < KD; ++t) { sv[wv][lane * KD + t] = lv[t]; si[wv][lane * KD + t] = li[t]; }
  }
  __syncthreads();
  if (active && lane == 0) {
    int ptr[32];
    for (int l = 0; l < 32; ++l) ptr[l] = 0;
    for (int t = 0; t < KD; ++t) {
      float best = 3.4e38f; int bl = 0;
      for (int l = 0; l < 32; ++l) {
        int p = ptr[l];
        if (p < KD) { float v = sv[wv][l * KD + p]; if (v < best) { best = v; bl = l; } }
      }
      int j = si[wv][bl * KD + ptr[bl]];
      ptr[bl]++;
      if (t % dil == 0) idxout[(size_t)row * k + t / dil] = j;
    }
  }
}

// gather neighbors, max-aggregate, write concat [f, max_j(f_j - f_i)] (rows x 2C)
__global__ void gather_max_k(const float* __restrict__ f, const int* __restrict__ idx,
                             float* __restrict__ cat, int rows, int N, int C, int k) {
  int wv = threadIdx.x >> 5, lane = threadIdx.x & 31;
  int row = blockIdx.x * 8 + wv;
  if (row >= rows) return;
  int b = row / N;
  const int* ip = idx + (size_t)row * k;
  int nb[16];
  for (int j = 0; j < k; ++j) nb[j] = ip[j];
  const float* fr = f + (size_t)row * C;
  float* cr = cat + (size_t)row * (2 * C);
  for (int c = lane; c < C; c += 32) {
    float fi = fr[c];
    float m = -3.4e38f;
    for (int j = 0; j < k; ++j) {
      float vj = f[((size_t)b * N + nb[j]) * C + c];
      m = fmaxf(m, vj - fi);
    }
    cr[c] = fi;
    cr[C + c] = m;
  }
}

// ---------------------------------------------------------------------------
// LDS-tiled transpose of flat (B,C,N) -> (B,N,C)
__global__ void transpose_cn_k(const float* __restrict__ in, float* __restrict__ outp,
                               int C, int N) {
  __shared__ float tile[32][33];
  int n0 = blockIdx.x * 32;
  int c0 = blockIdx.y * 32;
  int b  = blockIdx.z;
#pragma unroll
  for (int i = 0; i < 4; ++i) {
    int c = c0 + threadIdx.y + i * 8, n = n0 + threadIdx.x;
    if (c < C && n < N) tile[threadIdx.y + i * 8][threadIdx.x] = in[((size_t)b * C + c) * N + n];
  }
  __syncthreads();
#pragma unroll
  for (int i = 0; i < 4; ++i) {
    int n = n0 + threadIdx.y + i * 8, c = c0 + threadIdx.x;
    if (c < C && n < N) outp[((size_t)b * N + n) * C + c] = tile[threadIdx.x][threadIdx.y + i * 8];
  }
}

// ---------------------------------------------------------------------------
// Stem: direct 3x3 conv + bias + BN(+ReLU). Small fraction of total FLOPs.
__global__ void conv3x3_bn_k(const float* __restrict__ x, const float* __restrict__ w,
                             const float* __restrict__ bias, const float* __restrict__ g,
                             const float* __restrict__ bt, float* __restrict__ out,
                             int Bn, int Ci, int Hi, int Wi, int Co, int Ho, int Wo,
                             int stride, int doRelu) {
  int idx = blockIdx.x * blockDim.x + threadIdx.x;
  int total = Bn * Co * Ho * Wo;
  if (idx >= total) return;
  int wo = idx % Wo; int t = idx / Wo;
  int ho = t % Ho;  t /= Ho;
  int co = t % Co;  int b = t / Co;
  float acc = 0.f;
  for (int ci = 0; ci < Ci; ++ci) {
    const float* xp = x + ((size_t)(b * Ci + ci)) * Hi * Wi;
    const float* wp = w + ((size_t)(co * Ci + ci)) * 9;
#pragma unroll
    for (int kh = 0; kh < 3; ++kh) {
      int h = ho * stride - 1 + kh;
      if (h < 0 || h >= Hi) continue;
#pragma unroll
      for (int kw = 0; kw < 3; ++kw) {
        int ww = wo * stride - 1 + kw;
        if (ww < 0 || ww >= Wi) continue;
        acc += xp[h * Wi + ww] * wp[kh * 3 + kw];
      }
    }
  }
  float v = (acc + bias[co]) * (BN_INV * g[co]) + bt[co];
  if (doRelu) v = fmaxf(v, 0.f);
  out[idx] = v;
}

// h + pos_embed, then expmap0 over channel dim. One wave per (b, spatial n).
__global__ void expmap_pos_k(const float* __restrict__ h, const float* __restrict__ pos,
                             float* __restrict__ X, int Bn, int C, int N) {
  int wv = threadIdx.x >> 5, lane = threadIdx.x & 31;
  int row = blockIdx.x * 8 + wv;
  if (row >= Bn * N) return;
  int b = row / N, n = row % N;
  float u[16];
  float s2 = 0.f;
  int t = 0;
  for (int c = lane; c < C; c += 32, ++t) {
    float v = h[((size_t)b * C + c) * N + n] + pos[(size_t)c * N + n];
    u[t] = v;
    if (c > 0) s2 += v * v;
  }
  s2 = waveRedAdd(s2);
  float nrm = sqrtf(fmaxf(s2, 1e-8f));
  float ch = coshf(nrm), sh = sinhf(nrm) / nrm;
  t = 0;
  for (int c = lane; c < C; c += 32, ++t)
    X[((size_t)b * C + c) * N + n] = (c == 0) ? ch : sh * u[t];
}

__global__ void pool_mean_k(const float* __restrict__ X, float* __restrict__ pooled,
                            int rows, int N) {
  int wv = threadIdx.x >> 5, lane = threadIdx.x & 31;
  int row = blockIdx.x * 8 + wv;
  if (row >= rows) return;
  const float* xr = X + (size_t)row * N;
  float s = 0.f;
  for (int n = lane; n < N; n += 32) s += xr[n];
  s = waveRedAdd(s);
  if (lane == 0) pooled[row] = s / (float)N;
}

__global__ void classify_k(const float* __restrict__ pooled, const float* __restrict__ clsw,
                           const float* __restrict__ hb, float* __restrict__ out,
                           int Bn, int C, int NC) {
  int idx = blockIdx.x * blockDim.x + threadIdx.x;
  if (idx >= Bn * NC) return;
  int b = idx / NC, kc = idx % NC;
  const float* p  = pooled + (size_t)b * C;
  const float* wr = clsw + (size_t)kc * C;
  float acc = -p[0] * wr[0];          // signs = [-1, +1, +1, ...]
  for (int c = 1; c < C; ++c) acc += p[c] * wr[c];
  out[idx] = 2.f + 2.f * acc + hb[kc];
}

// ===========================================================================
extern "C" void kernel_launch(void* const* d_in, const int* in_sizes, int n_in,
                              void* d_out, int out_size, void* d_ws, size_t ws_size,
                              hipStream_t stream) {
  (void)in_sizes; (void)n_in; (void)out_size;
  constexpr int B = 2, C = 320, L = 6, Kk = 9, NCLS = 1000;
  constexpr int C1 = 160, H1 = 112, W1 = 112, H2 = 56, W2 = 56;
  constexpr int N = H2 * W2;        // 3136 tokens per image
  constexpr int MT = B * N;         // 6272 total tokens

  const float* x    = (const float*)d_in[0];
  const float* c1w  = (const float*)d_in[1];  const float* c1b = (const float*)d_in[2];
  const float* g1   = (const float*)d_in[3];  const float* bt1 = (const float*)d_in[4];
  const float* c2w  = (const float*)d_in[5];  const float* c2b = (const float*)d_in[6];
  const float* g2   = (const float*)d_in[7];  const float* bt2 = (const float*)d_in[8];
  const float* c3w  = (const float*)d_in[9];  const float* c3b = (const float*)d_in[10];
  const float* g3   = (const float*)d_in[11]; const float* bt3 = (const float*)d_in[12];
  const float* pos  = (const float*)d_in[13];
  // dict order: fc1_*, fc2_*, ffn_* (w1,b1,s1,w2,b2,s2 each), then gc_w/b/s, cls, head
  const float* fw1[3], *fb1[3], *fs1[3], *fw2[3], *fb2[3], *fs2[3];
  for (int j = 0; j < 3; ++j) {
    fw1[j] = (const float*)d_in[14 + 6 * j + 0];
    fb1[j] = (const float*)d_in[14 + 6 * j + 1];
    fs1[j] = (const float*)d_in[14 + 6 * j + 2];
    fw2[j] = (const float*)d_in[14 + 6 * j + 3];
    fb2[j] = (const float*)d_in[14 + 6 * j + 4];
    fs2[j] = (const float*)d_in[14 + 6 * j + 5];
  }
  const float* gcw  = (const float*)d_in[32];
  const float* gcb  = (const float*)d_in[33];
  const float* gcs  = (const float*)d_in[34];
  const float* clsw = (const float*)d_in[35];
  const float* hb   = (const float*)d_in[36];
  float* out = (float*)d_out;

  // ---- workspace bump allocator ----
  uintptr_t base = (uintptr_t)d_ws;
  auto alloc = [&](size_t bytes) -> void* {
    uintptr_t p = (base + 255) & ~(uintptr_t)255;
    base = p + bytes;
    return (void*)p;
  };
  _Float16* w16[6];   // fc1w1, fc1w2, fc2w1, fc2w2, ffnw1, ffnw2
  for (int j = 0; j < 6; ++j) w16[j] = (_Float16*)alloc((size_t)L * C * C * 2);
  _Float16* gc16 = (_Float16*)alloc((size_t)L * C * 2 * C * 2);
  float* h1   = (float*)alloc((size_t)B * C1 * H1 * W1 * 4);
  float* h2   = (float*)alloc((size_t)B * C  * H2 * W2 * 4);
  float* X    = (float*)alloc((size_t)MT * C * 4);     // flat (B,C,N)
  float* T    = (float*)alloc((size_t)MT * C * 4);     // flat (B,N,C)
  float* Y1   = (float*)alloc((size_t)MT * C * 4);
  float* Y2   = (float*)alloc((size_t)MT * C * 4);
  float* CAT  = (float*)alloc((size_t)MT * 2 * C * 4);
  _Float16* A16 = (_Float16*)alloc((size_t)MT * 2 * C * 2);
  float* dist = (float*)alloc((size_t)B * N * N * 4);  // 78.7 MB — fits in 192 MB L2
  float* sq   = (float*)alloc((size_t)MT * 4);
  int*   idxb = (int*)alloc((size_t)MT * Kk * 4);
  float* pooled = (float*)alloc((size_t)B * C * 4);
  if ((size_t)(base - (uintptr_t)d_ws) > ws_size) return;  // insufficient scratch

  auto cdiv = [](long a, long b) { return (int)((a + b - 1) / b); };
  auto cvtK = [&](const float* s, _Float16* d, size_t n, int relu) {
    cvt_f16_k<<<cdiv((long)n, 256), 256, 0, stream>>>(s, d, n, relu);
  };
  auto gemmK = [&](const _Float16* A, const _Float16* Bw, const float* bias,
                   const float* sqv, float* O, int M, int Nn, int K, int mode) {
    int tiles = (M / 16) * (Nn / 16);
    gemm_wmma_f16<<<cdiv(tiles, 8), 256, 0, stream>>>(A, Bw, bias, sqv, O, M, Nn, K, mode, tiles);
  };
  auto lorK = [&](float* Y, int rows, const float* s) {
    lorentz_rows_k<<<cdiv(rows, 8), 256, 0, stream>>>(Y, rows, C, s);
  };
  auto addK = [&](float* a, const float* b, size_t n) {
    add_inplace_k<<<cdiv((long)n, 256), 256, 0, stream>>>(a, b, n);
  };
  auto transK = [&](const float* in, float* o) {  // flat (B,C,N) -> (B,N,C)
    dim3 grid(cdiv(N, 32), cdiv(C, 32), B), blk(32, 8);
    transpose_cn_k<<<grid, blk, 0, stream>>>(in, o, C, N);
  };
  // ffn(IN) = LL(LL(IN,w1),w2) + IN ; IN,OUT,TMP all distinct (B,N,C) buffers
  auto ffnK = [&](float* IN, float* OUT, float* TMP,
                  const _Float16* w1, const float* b1, const float* s1,
                  const _Float16* w2, const float* b2, const float* s2) {
    cvtK(IN, A16, (size_t)MT * C, 1);
    gemmK(A16, w1, b1, nullptr, TMP, MT, C, C, 0);
    lorK(TMP, MT, s1);
    cvtK(TMP, A16, (size_t)MT * C, 1);
    gemmK(A16, w2, b2, nullptr, OUT, MT, C, C, 0);
    lorK(OUT, MT, s2);
    addK(OUT, IN, (size_t)MT * C);
  };

  // ---- weights -> f16 (every call; no cached state) ----
  cvtK(fw1[0], w16[0], (size_t)L * C * C, 0);
  cvtK(fw2[0], w16[1], (size_t)L * C * C, 0);
  cvtK(fw1[1], w16[2], (size_t)L * C * C, 0);
  cvtK(fw2[1], w16[3], (size_t)L * C * C, 0);
  cvtK(fw1[2], w16[4], (size_t)L * C * C, 0);
  cvtK(fw2[2], w16[5], (size_t)L * C * C, 0);
  cvtK(gcw, gc16, (size_t)L * C * 2 * C, 0);

  // ---- stem ----
  conv3x3_bn_k<<<cdiv((long)B * C1 * H1 * W1, 256), 256, 0, stream>>>(
      x, c1w, c1b, g1, bt1, h1, B, 3, 224, 224, C1, H1, W1, 2, 1);
  conv3x3_bn_k<<<cdiv((long)B * C * H2 * W2, 256), 256, 0, stream>>>(
      h1, c2w, c2b, g2, bt2, h2, B, C1, H1, W1, C, H2, W2, 2, 1);
  conv3x3_bn_k<<<cdiv((long)B * C * H2 * W2, 256), 256, 0, stream>>>(
      h2, c3w, c3b, g3, bt3, Y1, B, C, H2, W2, C, H2, W2, 1, 0);
  expmap_pos_k<<<cdiv(MT, 8), 256, 0, stream>>>(Y1, pos, X, B, C, N);

  // ---- 6 graph blocks ----
  for (int i = 0; i < L; ++i) {
    int dil = (i / 4) + 1;                       // DILS = 1,1,1,1,2,2
    const _Float16* l_fc1w1 = w16[0] + (size_t)i * C * C;
    const _Float16* l_fc1w2 = w16[1] + (size_t)i * C * C;
    const _Float16* l_fc2w1 = w16[2] + (size_t)i * C * C;
    const _Float16* l_fc2w2 = w16[3] + (size_t)i * C * C;
    const _Float16* l_ffnw1 = w16[4] + (size_t)i * C * C;
    const _Float16* l_ffnw2 = w16[5] + (size_t)i * C * C;
    const _Float16* l_gcw   = gc16  + (size_t)i * C * 2 * C;

    transK(X, T);                                              // t0 (B,N,C)
    ffnK(T, Y2, Y1, l_fc1w1, fb1[0] + i * C, fs1[0] + i,
                    l_fc1w2, fb2[0] + i * C, fs2[0] + i);      // t1 = Y2
    transK(Y2, T);                                             // f  (reshape+transpose of t1)

    // graph conv: Gram via WMMA, dist fused in epilogue (stays resident in L2)
    row_sqnorm_k<<<cdiv(MT, 8), 256, 0, stream>>>(T, sq, MT, C);
    cvtK(T, A16, (size_t)MT * C, 0);                           // raw f (no relu) for distances
    for (int b = 0; b < B; ++b)
      gemmK(A16 + (size_t)b * N * C, A16 + (size_t)b * N * C, nullptr,
            sq + (size_t)b * N, dist + (size_t)b * N * N, N, N, C, 1);
    if (dil == 1)
      topk_rows_k<9><<<cdiv(MT, 8), 256, 0, stream>>>(dist, idxb, MT, N, 1, Kk);
    else
      topk_rows_k<18><<<cdiv(MT, 8), 256, 0, stream>>>(dist, idxb, MT, N, 2, Kk);
    gather_max_k<<<cdiv(MT, 8), 256, 0, stream>>>(T, idxb, CAT, MT, N, C, Kk);
    cvtK(CAT, A16, (size_t)MT * 2 * C, 1);
    gemmK(A16, l_gcw, gcb + i * C, nullptr, Y1, MT, C, 2 * C, 0);
    lorK(Y1, MT, gcs + i);                                     // y = Y1 (double-transpose elided)

    ffnK(Y1, Y2, T, l_fc2w1, fb1[1] + i * C, fs1[1] + i,
                    l_fc2w2, fb2[1] + i * C, fs2[1] + i);      // t2 = Y2
    addK(X, Y2, (size_t)MT * C);                               // flat add = reshape + shortcut
    transK(X, T);
    ffnK(T, Y2, Y1, l_ffnw1, fb1[2] + i * C, fs1[2] + i,
                    l_ffnw2, fb2[2] + i * C, fs2[2] + i);      // t3 = Y2
    float* tmp = X; X = Y2; Y2 = tmp;                          // block output (flat reinterp)
  }

  // ---- head ----
  pool_mean_k<<<cdiv(B * C, 8), 256, 0, stream>>>(X, pooled, B * C, N);
  classify_k<<<cdiv(B * NCLS, 256), 256, 0, stream>>>(pooled, clsw, hb, out, B, C, NCLS);
}
